// W8Linear_43336220016658
// MI455X (gfx1250) — matrix operand order repaired and checked by reading, not compiled, not run on hardware
//
#include <hip/hip_runtime.h>
#include <stdint.h>

typedef __attribute__((ext_vector_type(16))) _Float16 v16h;
typedef __attribute__((ext_vector_type(8)))  _Float16 v8h;
typedef __attribute__((ext_vector_type(4)))  _Float16 v4h;
typedef __attribute__((ext_vector_type(8)))  float    v8f;
typedef __attribute__((ext_vector_type(4)))  float    v4f;
typedef __attribute__((ext_vector_type(4)))  int      v4i;

#define M_TOTAL 8192      // B*S = 4*2048
#define N_TOTAL 11008     // OUT_F
#define K_TOTAL 4096      // IN_F

#define BM 128
#define BN 256
#define BK 64
#define LDS_K 72          // BK + 8 f16 pad -> 144B row stride, 16B aligned, bank-spread

#define A_STAGE (BM * LDS_K)   // f16 elements per A stage
#define B_STAGE (BN * LDS_K)   // f16 elements per B stage

union HV { v16h v; v8h h[2]; };

__global__ __launch_bounds__(256)
void w8linear_wmma(const float* __restrict__ x,      // [M,K] fp16 values widened to f32
                   const int*   __restrict__ qw,     // [N,K] int8 values widened to i32
                   const float* __restrict__ scale,  // [N]
                   const float* __restrict__ bias,   // [N]
                   float*       __restrict__ out)    // [M,N]
{
    // Double-buffered stages: 2*(18KB + 36KB) = 108KB of the 320KB WGP LDS.
    __shared__ __align__(16) _Float16 As[2 * A_STAGE];
    __shared__ __align__(16) _Float16 Bs[2 * B_STAGE];

    const int tid  = threadIdx.x;
    const int lane = tid & 31;
    const int wave = tid >> 5;
    const int wm   = wave >> 2;   // 0..1 : 64 output rows each
    const int wn   = wave & 3;    // 0..3 : 64 output cols each

    const int mBlk = blockIdx.y * BM;
    const int nBlk = blockIdx.x * BN;

    // LDS fill mapping: 16 threads per row, 4 elements per thread per pass.
    const int fr = tid >> 4;          // 0..15
    const int fc = (tid & 15) << 2;   // 0,4,...,60

    // WMMA operand addressing (CDNA5 16-bit A/B VGPR layouts, wave32):
    const int laneMod = lane & 15;
    const int laneHi  = lane >> 4;        // 0 or 1
    const int aKlo    = laneHi << 3;      // A lane: K = {klo..klo+7} U {klo+16..klo+23}
    const int bKlo    = laneHi << 3;      // B lane: K = {klo..klo+7} U {klo+16..klo+23}

    v8f acc[4][4];
#pragma unroll
    for (int mi = 0; mi < 4; ++mi)
#pragma unroll
        for (int ni = 0; ni < 4; ++ni)
            acc[mi][ni] = (v8f)(0.0f);

    // ---------------- prologue fill of stage 0 ----------------
#pragma unroll
    for (int p = 0; p < 8; ++p) {
        const int row = fr + p * 16;
        v4f v = *(const v4f*)&x[(size_t)(mBlk + row) * K_TOTAL + fc];
        v4h t;
        t.x = (_Float16)v.x; t.y = (_Float16)v.y;
        t.z = (_Float16)v.z; t.w = (_Float16)v.w;
        *(v4h*)&As[row * LDS_K + fc] = t;
    }
#pragma unroll
    for (int p = 0; p < 16; ++p) {
        const int row = fr + p * 16;
        v4i w = *(const v4i*)&qw[(size_t)(nBlk + row) * K_TOTAL + fc];
        v4h t;
        t.x = (_Float16)w.x; t.y = (_Float16)w.y;
        t.z = (_Float16)w.z; t.w = (_Float16)w.w;
        *(v4h*)&Bs[row * LDS_K + fc] = t;
    }
    __syncthreads();

    int buf = 0;
#pragma unroll 1
    for (int kb = 0; kb < K_TOTAL; kb += BK) {
        const int nkb = kb + BK;

        // ---- fill the other stage with the next K slab (overlaps WMMA below) ----
        if (nkb < K_TOTAL) {
            __builtin_prefetch(&x [(size_t)(mBlk + (tid >> 1)) * K_TOTAL + nkb + BK + (tid & 1) * 32], 0, 1);
            __builtin_prefetch(&qw[(size_t)(nBlk + (tid >> 1)) * K_TOTAL + nkb + BK + (tid & 1) * 32], 0, 1);

            const int aOff = (buf ^ 1) * A_STAGE;
            const int bOff = (buf ^ 1) * B_STAGE;
#pragma unroll
            for (int p = 0; p < 8; ++p) {
                const int row = fr + p * 16;
                v4f v = *(const v4f*)&x[(size_t)(mBlk + row) * K_TOTAL + nkb + fc];
                v4h t;
                t.x = (_Float16)v.x; t.y = (_Float16)v.y;
                t.z = (_Float16)v.z; t.w = (_Float16)v.w;
                *(v4h*)&As[aOff + row * LDS_K + fc] = t;
            }
#pragma unroll
            for (int p = 0; p < 16; ++p) {
                const int row = fr + p * 16;
                v4i w = *(const v4i*)&qw[(size_t)(nBlk + row) * K_TOTAL + nkb + fc];
                v4h t;
                t.x = (_Float16)w.x; t.y = (_Float16)w.y;
                t.z = (_Float16)w.z; t.w = (_Float16)w.w;
                *(v4h*)&Bs[bOff + row * LDS_K + fc] = t;
            }
        }

        // ---- compute from the current stage: 2 k-steps x 16 WMMAs ----
        const int aOffC = buf * A_STAGE;
        const int bOffC = buf * B_STAGE;
#pragma unroll
        for (int ks = 0; ks < BK; ks += 32) {
            // Preload ALL fragments for this k-step so the 16 ds_load_b128 are
            // in flight together and the 16 WMMAs drain without per-quartet stalls.
            HV a[4], b[4];
#pragma unroll
            for (int ni = 0; ni < 4; ++ni) {
                const int col = wn * 64 + ni * 16 + laneMod;
                b[ni].h[0] = *(const v8h*)&Bs[bOffC + col * LDS_K + ks + bKlo];
                b[ni].h[1] = *(const v8h*)&Bs[bOffC + col * LDS_K + ks + bKlo + 16];
            }
#pragma unroll
            for (int mi = 0; mi < 4; ++mi) {
                const int row = wm * 64 + mi * 16 + laneMod;
                a[mi].h[0] = *(const v8h*)&As[aOffC + row * LDS_K + ks + aKlo];
                a[mi].h[1] = *(const v8h*)&As[aOffC + row * LDS_K + ks + aKlo + 16];
            }
#pragma unroll
            for (int mi = 0; mi < 4; ++mi)
#pragma unroll
                for (int ni = 0; ni < 4; ++ni)
                    acc[mi][ni] = __builtin_amdgcn_wmma_f32_16x16x32_f16(
                        /*neg_a=*/false, a[mi].v,
                        /*neg_b=*/false, b[ni].v,
                        /*c_mod=*/(short)0, acc[mi][ni],
                        /*reuse_a=*/false, /*reuse_b=*/false);
        }
        __syncthreads();
        buf ^= 1;
    }

    // ---------------- epilogue: y = acc*scale[col] + bias[col] ----------------
#pragma unroll
    for (int mi = 0; mi < 4; ++mi) {
#pragma unroll
        for (int ni = 0; ni < 4; ++ni) {
            const int col = nBlk + wn * 64 + ni * 16 + laneMod;
            const float sc = scale[col];
            const float bi = bias[col];
            const int row0 = mBlk + wm * 64 + mi * 16 + (laneHi << 3);
#pragma unroll
            for (int g = 0; g < 8; ++g) {
                // fp16 rounding to match the reference output dtype, stored widened
                out[(size_t)(row0 + g) * N_TOTAL + col] =
                    (float)(_Float16)(acc[mi][ni][g] * sc + bi);
            }
        }
    }
}

extern "C" void kernel_launch(void* const* d_in, const int* in_sizes, int n_in,
                              void* d_out, int out_size, void* d_ws, size_t ws_size,
                              hipStream_t stream) {
    const float* x     = (const float*)d_in[0];
    const int*   qw    = (const int*)  d_in[1];
    const float* scale = (const float*)d_in[2];
    const float* bias  = (const float*)d_in[3];
    float*       out   = (float*)d_out;

    dim3 grid(N_TOTAL / BN, M_TOTAL / BM);   // 43 x 64, exact tiling
    w8linear_wmma<<<grid, 256, 0, stream>>>(x, qw, scale, bias, out);
}
